// BilinAndFwdComboVecComp_66571993088480
// MI455X (gfx1250) — matrix operand order repaired and checked by reading, not compiled, not run on hardware
//
#include <hip/hip_runtime.h>
#include <hip/hip_bf16.h>
#include <math.h>

// Problem sizes (fixed by the reference)
#define Bn 8
#define Sn 256
#define Cn 256
#define Vn 64

// LDS pitch for a staged B panel: 264 halves = 528 B = 132 dwords.
// 132 % 64 banks = 4-bank stride between columns -> B-tile reads are ~2-way
// bank conflicted instead of fully serialized (pitch 256 would alias bank 0).
#define PITCH 264

typedef __attribute__((ext_vector_type(16))) __bf16 v16bf;
typedef __attribute__((ext_vector_type(8)))  float  v8f;

union TileBF {          // one WMMA 16-bit operand: 8 VGPRs = 16 bf16 halves
  v16bf v;
  uint4 q[2];
};

__device__ __forceinline__ unsigned short f2bf(float f) {
  unsigned int u = __float_as_uint(f);
  u += 0x7FFFu + ((u >> 16) & 1u);   // round-to-nearest-even
  return (unsigned short)(u >> 16);
}

// ---- CDNA5 async global->LDS copy (ASYNCcnt-tracked, ISA ch.10 / 08_async_tensor)
__device__ __forceinline__ void async_copy_b128(unsigned lds_off, const void* gptr) {
  asm volatile("global_load_async_to_lds_b128 %0, %1, off"
               :: "v"(lds_off), "v"(gptr) : "memory");
}
__device__ __forceinline__ void async_wait0() {
#if __has_builtin(__builtin_amdgcn_s_wait_asynccnt)
  __builtin_amdgcn_s_wait_asynccnt(0);
#else
  asm volatile("s_wait_asynccnt 0x0" ::: "memory");
#endif
}

// ---------------------------------------------------------------- prep: ctx -> bf16
__global__ void k_cvt_ctx(const float* __restrict__ ctx,
                          unsigned short* __restrict__ dst, int n) {
  int i = blockIdx.x * blockDim.x + threadIdx.x;
  if (i < n) dst[i] = f2bf(ctx[i]);
}

// ---------------------------------------------------------------- prep: W(C,C,V) -> W_T bf16 [(v*C+j)][i]
__global__ void k_build_wt(const float* __restrict__ W,
                           unsigned short* __restrict__ wt) {
  int idx = blockIdx.x * blockDim.x + threadIdx.x;     // [0, C*V*C)
  if (idx >= Cn * Vn * Cn) return;
  int i    = idx % Cn;
  int ncol = idx / Cn;        // ncol = v*C + j
  int j    = ncol % Cn;
  int v    = ncol / Cn;
  wt[idx] = f2bf(W[(size_t)(i * Cn + j) * Vn + v]);
}

// ---------------------------------------------------------------- row terms R,Q (f32)
// R[b,z,v] = sum_c ctx[b,z,c]*(lin1_w+lindiff_w)[v,c] + all biases
// Q[b,s,v] = sum_c ctx[b,s,c]*(lin2_w-lindiff_w)[v,c]
__global__ void k_rq(const float* __restrict__ ctx,
                     const float* __restrict__ bias,
                     const float* __restrict__ lin1_w, const float* __restrict__ lin1_b,
                     const float* __restrict__ lin2_w, const float* __restrict__ lin2_b,
                     const float* __restrict__ linmul_b,
                     const float* __restrict__ lindiff_w, const float* __restrict__ lindiff_b,
                     float* __restrict__ R, float* __restrict__ Q) {
  __shared__ float row[Cn];
  const int bs = blockIdx.x;           // b*S + row
  const int v  = threadIdx.x;          // 0..63
  for (int c = v; c < Cn; c += blockDim.x) row[c] = ctx[(size_t)bs * Cn + c];
  __syncthreads();
  float a1 = 0.f, a2 = 0.f;
  for (int c = 0; c < Cn; ++c) {
    const float x  = row[c];
    const float wd = lindiff_w[v * Cn + c];
    a1 = fmaf(x, lin1_w[v * Cn + c] + wd, a1);
    a2 = fmaf(x, lin2_w[v * Cn + c] - wd, a2);
  }
  R[(size_t)bs * Vn + v] = a1 + bias[v] + lin1_b[v] + lin2_b[v] + linmul_b[v] + lindiff_b[v];
  Q[(size_t)bs * Vn + v] = a2;
}

// ---------------------------------------------------------------- stage 1 (WMMA bf16, LDS-staged B)
// E_T[b,s][v*C+j] = bf16( sum_i ctx[b,s,i]*W[i,j,v] + ctx[b,s,j]*linmul_w[v,j] )
// Block = 512 thr (16 waves) = one 64-col panel of W_T for one b; wave w -> s-tile w.
// B panel (64 cols x 256 K, bf16 = 32 KB) async-staged once into LDS.
__global__ void __launch_bounds__(512)
k_stage1(const unsigned short* __restrict__ ctxb,   // bf16 (B,S,C)
         const unsigned short* __restrict__ wt,     // bf16 (C*V, C)
         const float* __restrict__ ctx,             // f32  (B,S,C)
         const float* __restrict__ linmul_w,        // f32  (V,C)
         unsigned short* __restrict__ Et)           // bf16 (B*S, C*V)
{
  __shared__ __align__(16) unsigned short sB[64 * PITCH];   // ~33 KB

  const int tid  = threadIdx.x;
  const int wave = tid >> 5;
  const int lane = tid & 31;
  const int b    = blockIdx.x >> 8;                 // 256 panels per b
  const int n0   = (blockIdx.x & 255) * 64;

  // async-stage B panel: 64 cols x 512 B, pitched. 2048 16-byte chunks / 512 thr.
  const unsigned sbase = (unsigned)(size_t)&sB[0];
#pragma unroll
  for (int it = 0; it < 4; ++it) {
    const int idx = it * 512 + tid;
    const int col = idx >> 5;                       // 0..63
    const int ch  = idx & 31;                       // 32 chunks of 8 halves
    async_copy_b128(sbase + (unsigned)(col * PITCH + ch * 8) * 2,
                    wt + (size_t)(n0 + col) * Cn + ch * 8);
  }
  async_wait0();
  __syncthreads();

  const int mr = lane & 15;                         // A row / B,D column lane
  const int g  = lane >> 4;                         // K-half / M-half select
  const int s0 = wave * 16;

  v8f acc[4] = {};
  const unsigned short* arow = ctxb + (size_t)(b * Sn + s0 + mr) * Cn;

  for (int kb = 0; kb < Cn; kb += 32) {
    TileBF a;                                       // A 16x32: K = kb+8g..+7, kb+16+8g..+7
    a.q[0] = *(const uint4*)(arow + kb + 8 * g);
    a.q[1] = *(const uint4*)(arow + kb + 16 + 8 * g);
#pragma unroll
    for (int t = 0; t < 4; ++t) {
      const unsigned short* bp = &sB[(t * 16 + mr) * PITCH + kb + 8 * g];
      TileBF bm;                                    // B 32x16: K = kb+8g..+7, kb+16+8g..+7
      bm.q[0] = *(const uint4*)(bp);
      bm.q[1] = *(const uint4*)(bp + 16);
      acc[t] = __builtin_amdgcn_wmma_f32_16x16x32_bf16(
          false, a.v, false, bm.v, (short)0, acc[t], false, false);
    }
  }

#pragma unroll
  for (int t = 0; t < 4; ++t) {
    const int col = n0 + t * 16 + mr;               // = v*C + j
    const int j   = col & (Cn - 1);
    const int v   = col >> 8;                       // C == 256
    const float wm = linmul_w[v * Cn + j];
#pragma unroll
    for (int r = 0; r < 8; ++r) {
      const int s = s0 + r + 8 * g;                 // C/D layout: m = r + 8*(lane>>4)
      const float e = acc[t][r] + ctx[(size_t)(b * Sn + s) * Cn + j] * wm;
      Et[(size_t)(b * Sn + s) * (Cn * Vn) + col] = f2bf(e);
    }
  }
}

// ---------------------------------------------------------------- stage 2 (WMMA bf16, LDS-staged E) + epilogue
// out[b,s,z,v] = tanh( sum_j ctx[b,z,j]*E_T[b,s][v*C+j] + R[b,z,v] + Q[b,s,v] )
// Block = 512 thr (16 waves) = one (b,s); E[b,s] (32 KB) async-staged into LDS;
// wave w computes z-tile w (16 z rows x 64 v).
__global__ void __launch_bounds__(512)
k_stage2(const unsigned short* __restrict__ ctxb,   // bf16 (B,S,C)
         const unsigned short* __restrict__ Et,     // bf16 (B*S, C*V) N-major
         const float* __restrict__ R,
         const float* __restrict__ Q,
         float* __restrict__ out)                   // f32 (B,S,S,V)
{
  __shared__ __align__(16) unsigned short sE[64 * PITCH];   // ~33 KB

  const int tid  = threadIdx.x;
  const int wave = tid >> 5;
  const int lane = tid & 31;
  const int bs   = blockIdx.x;                      // b*S + s
  const int b    = bs >> 8;                         // S == 256

  const unsigned short* eb = Et + (size_t)bs * (Cn * Vn);
  const unsigned sbase = (unsigned)(size_t)&sE[0];
#pragma unroll
  for (int it = 0; it < 4; ++it) {
    const int idx = it * 512 + tid;
    const int col = idx >> 5;                       // v = 0..63
    const int ch  = idx & 31;
    async_copy_b128(sbase + (unsigned)(col * PITCH + ch * 8) * 2,
                    eb + (size_t)col * Cn + ch * 8);
  }
  async_wait0();
  __syncthreads();

  const int mr = lane & 15;
  const int g  = lane >> 4;
  const int z0 = wave * 16;

  const unsigned short* arow = ctxb + (size_t)(b * Sn + z0 + mr) * Cn;

  v8f acc[4] = {};
  for (int kb = 0; kb < Cn; kb += 32) {
    TileBF a;
    a.q[0] = *(const uint4*)(arow + kb + 8 * g);
    a.q[1] = *(const uint4*)(arow + kb + 16 + 8 * g);
#pragma unroll
    for (int t = 0; t < 4; ++t) {
      const unsigned short* bp = &sE[(t * 16 + mr) * PITCH + kb + 8 * g];
      TileBF bm;
      bm.q[0] = *(const uint4*)(bp);
      bm.q[1] = *(const uint4*)(bp + 16);
      acc[t] = __builtin_amdgcn_wmma_f32_16x16x32_bf16(
          false, a.v, false, bm.v, (short)0, acc[t], false, false);
    }
  }

  float qv[4];
#pragma unroll
  for (int t = 0; t < 4; ++t) qv[t] = Q[(size_t)bs * Vn + t * 16 + mr];

#pragma unroll
  for (int t = 0; t < 4; ++t) {
    const int v = t * 16 + mr;
#pragma unroll
    for (int r = 0; r < 8; ++r) {
      const int z = z0 + r + 8 * g;
      const float sum = acc[t][r] + R[(size_t)(b * Sn + z) * Vn + v] + qv[t];
      out[((size_t)bs * Sn + z) * Vn + v] = tanhf(sum);
    }
  }
}

// ---------------------------------------------------------------- launch
extern "C" void kernel_launch(void* const* d_in, const int* in_sizes, int n_in,
                              void* d_out, int out_size, void* d_ws, size_t ws_size,
                              hipStream_t stream) {
  (void)in_sizes; (void)n_in; (void)out_size; (void)ws_size;

  const float* ctx       = (const float*)d_in[0];
  const float* W         = (const float*)d_in[1];
  const float* bias      = (const float*)d_in[2];
  const float* lin1_w    = (const float*)d_in[3];
  const float* lin1_b    = (const float*)d_in[4];
  const float* lin2_w    = (const float*)d_in[5];
  const float* lin2_b    = (const float*)d_in[6];
  const float* linmul_w  = (const float*)d_in[7];
  const float* linmul_b  = (const float*)d_in[8];
  const float* lindiff_w = (const float*)d_in[9];
  const float* lindiff_b = (const float*)d_in[10];

  // workspace layout (bytes):
  //   ctx_bf16 : 1 MiB | W_T bf16 : 8 MiB | E_T bf16 : 64 MiB | R : 512 KiB | Q : 512 KiB
  char* ws = (char*)d_ws;
  unsigned short* ctxb = (unsigned short*)(ws);
  unsigned short* wt   = (unsigned short*)(ws + (1u << 20));
  unsigned short* Et   = (unsigned short*)(ws + (1u << 20) + (8u << 20));
  float* R             = (float*)(ws + (1u << 20) + (8u << 20) + (64u << 20));
  float* Q             = (float*)(ws + (1u << 20) + (8u << 20) + (64u << 20) + (512u << 10));

  k_cvt_ctx <<<(Bn * Sn * Cn) / 256, 256, 0, stream>>>(ctx, ctxb, Bn * Sn * Cn);
  k_build_wt<<<(Cn * Vn * Cn) / 256, 256, 0, stream>>>(W, wt);
  k_rq      <<<Bn * Sn, Vn, 0, stream>>>(ctx, bias, lin1_w, lin1_b, lin2_w, lin2_b,
                                         linmul_b, lindiff_w, lindiff_b, R, Q);
  k_stage1  <<<Bn * 256, 512, 0, stream>>>(ctxb, wt, ctx, linmul_w, Et);
  k_stage2  <<<Bn * Sn, 512, 0, stream>>>(ctxb, Et, R, Q, (float*)d_out);
}
